// PartialGNNWithHead_6923487281995
// MI455X (gfx1250) — matrix operand order repaired and checked by reading, not compiled, not run on hardware
//
#include <hip/hip_runtime.h>
#include <math.h>

#define NNODES   19385
#define NEDGES   1200000
#define GDIM     256
#define HDIM     512
#define UDIM     2048      // HDIM * EXPAND
#define NCLS     3
#define NGENES   6640
#define NRES     6
#define BATCH    2048

typedef __attribute__((ext_vector_type(16))) __bf16 v16bf;
typedef __attribute__((ext_vector_type(8)))  float  v8f;

__device__ __forceinline__ unsigned short f2bf(float f) {
  unsigned int u = __builtin_bit_cast(unsigned int, f);
  u += 0x7FFFu + ((u >> 16) & 1u);          // round-to-nearest-even
  return (unsigned short)(u >> 16);
}
__device__ __forceinline__ unsigned int f2bf_pk(float lo, float hi) {
  return (unsigned int)f2bf(lo) | ((unsigned int)f2bf(hi) << 16);
}
__device__ __forceinline__ float gelu_exact(float x) {
  return 0.5f * x * (1.0f + erff(x * 0.7071067811865476f));
}

// CDNA5 async global->LDS copy (ASYNCcnt-tracked, no VGPR round-trip).
// VDST operand = LDS byte offset (generic LDS pointer low 32 bits per aperture rules).
__device__ __forceinline__ void async_copy_b128(unsigned lds_off, const void* gaddr) {
  asm volatile("global_load_async_to_lds_b128 %0, %1, off"
               :: "v"(lds_off), "v"((unsigned long long)(size_t)gaddr)
               : "memory");
}
__device__ __forceinline__ void wait_async0() {
#if defined(__has_builtin) && __has_builtin(__builtin_amdgcn_s_wait_asynccnt)
  __builtin_amdgcn_s_wait_asynccnt(0);
#else
  asm volatile("s_wait_asynccnt 0x0" ::: "memory");
#endif
}

// ---------------- utility kernels ----------------

__global__ __launch_bounds__(256) void cvt_f32_bf16(const float* __restrict__ s,
                                                    unsigned short* __restrict__ d, int n) {
  int i = blockIdx.x * 256 + threadIdx.x;
  if (i < n) d[i] = f2bf(s[i]);
}

// src [R,C] row-major f32 -> dst[c*dstStride + dstOff + r] = bf16(src[r*C + c])
// (i.e. transposed, N-major weights with optional K-offset for stacked weights)
__global__ __launch_bounds__(256) void cvt_tr_bf16(const float* __restrict__ s,
                                                   unsigned short* __restrict__ d,
                                                   int R, int C, int dstStride, int dstOff) {
  int i = blockIdx.x * 256 + threadIdx.x;
  if (i < R * C) {
    int r = i % R, c = i / R;
    d[(size_t)c * dstStride + dstOff + r] = f2bf(s[(size_t)r * C + c]);
  }
}

__global__ __launch_bounds__(256) void build_flags(const int* __restrict__ idx,
                                                   int* __restrict__ flags) {
  int i = blockIdx.x * 256 + threadIdx.x;
  if (i < BATCH) { int n = idx[i]; if (n >= 0) flags[n] = 1; }
}

// one wave per edge; only edges whose dst is a queried node contribute
__global__ __launch_bounds__(256) void edge_scatter(const int* __restrict__ ei,
                                                    const float* __restrict__ ew,
                                                    const float* __restrict__ frozen,
                                                    const int* __restrict__ flags,
                                                    float* __restrict__ aggr) {
  int e = blockIdx.x * 8 + (threadIdx.x >> 5);
  if (e >= NEDGES) return;
  int dst = ei[NEDGES + e];
  if (!flags[dst]) return;
  int lane = threadIdx.x & 31;
  int src = ei[e];
  float w = ew[e];
  const float* xr = frozen + (size_t)src * GDIM;
  float* ar = aggr + (size_t)dst * GDIM;
  #pragma unroll
  for (int i = 0; i < 2; ++i) {
    int c = lane * 4 + i * 128;
    float4 v = *(const float4*)(xr + c);
    atomicAdd(ar + c + 0, v.x * w);
    atomicAdd(ar + c + 1, v.y * w);
    atomicAdd(ar + c + 2, v.z * w);
    atomicAdd(ar + c + 3, v.w * w);
  }
}

// Acat[b, 0:256] = frozen[idx[b]] ; Acat[b, 256:512] = aggr[idx[b]] ; zeros if idx<0
__global__ __launch_bounds__(256) void gather_rows(const int* __restrict__ idx,
                                                   const float* __restrict__ frozen,
                                                   const float* __restrict__ aggr,
                                                   float* __restrict__ Acat) {
  int t = blockIdx.x * 256 + threadIdx.x;   // BATCH*64 threads
  int b = t >> 6, c = (t & 63) << 2;
  int n = idx[b];
  float4 xv = {0.f,0.f,0.f,0.f}, av = {0.f,0.f,0.f,0.f};
  if (n >= 0) {
    xv = *(const float4*)(frozen + (size_t)n * GDIM + c);
    av = *(const float4*)(aggr   + (size_t)n * GDIM + c);
  }
  *(float4*)(Acat + (size_t)b * (2*GDIM) + c)        = xv;
  *(float4*)(Acat + (size_t)b * (2*GDIM) + GDIM + c) = av;
}

__global__ __launch_bounds__(256) void fallback_fixup(const int* __restrict__ idx,
                                                      const float* __restrict__ fb,
                                                      float* __restrict__ P) {
  int t = blockIdx.x * 256 + threadIdx.x;   // BATCH*64
  int b = t >> 6, c = (t & 63) << 2;
  if (idx[b] < 0)
    *(float4*)(P + (size_t)b * GDIM + c) = *(const float4*)(fb + c);
}

// rows of 512, one wave per row, 8 rows per block
__global__ __launch_bounds__(256) void layernorm512(const float* __restrict__ X,
                                                    float* __restrict__ Y,
                                                    const float* __restrict__ g,
                                                    const float* __restrict__ b) {
  const int row  = blockIdx.x * 8 + (threadIdx.x >> 5);
  const int lane = threadIdx.x & 31;
  const float* x = X + (size_t)row * HDIM;
  float4 v[4];
  float s = 0.f, ss = 0.f;
  #pragma unroll
  for (int i = 0; i < 4; ++i) {
    v[i] = *(const float4*)(x + lane * 4 + i * 128);
    s  += v[i].x + v[i].y + v[i].z + v[i].w;
    ss += v[i].x*v[i].x + v[i].y*v[i].y + v[i].z*v[i].z + v[i].w*v[i].w;
  }
  #pragma unroll
  for (int off = 16; off > 0; off >>= 1) {
    s  += __shfl_xor(s,  off, 32);
    ss += __shfl_xor(ss, off, 32);
  }
  const float mu  = s * (1.f / 512.f);
  const float var = ss * (1.f / 512.f) - mu * mu;
  const float inv = rsqrtf(var + 1e-5f);
  float* y = Y + (size_t)row * HDIM;
  #pragma unroll
  for (int i = 0; i < 4; ++i) {
    const int c = lane * 4 + i * 128;
    float4 o;
    o.x = (v[i].x - mu) * inv * g[c+0] + b[c+0];
    o.y = (v[i].y - mu) * inv * g[c+1] + b[c+1];
    o.z = (v[i].z - mu) * inv * g[c+2] + b[c+2];
    o.w = (v[i].w - mu) * inv * g[c+3] + b[c+3];
    *(float4*)(y + c) = o;
  }
}

// ---------------- bf16 WMMA GEMM ----------------
// C[M,N] = epilogue( A_f32[M,K] @ B^T )  with B stored N-major: Bw[n*K + k].
// 256 threads = 8 waves (2 M x 4 N), block tile 64x128, wave tile 32x32, K-step 32.
// A: register-prefetch (needs f32->bf16 pack). B: async global->LDS copies into a
// double-buffered tile (ASYNCcnt), overlapping the WMMAs of the current slab.

#define BM 64
#define BN 128
#define BK 32
#define LDSK 40   // padded LDS row stride (ushorts) -> 80B rows, 16B-aligned fragments

template<bool GELU, bool HAS_BIAS, bool HAS_RES>
__global__ __launch_bounds__(256) void gemm_bf16_wmma(
    const float* __restrict__ A, const unsigned short* __restrict__ Bw,
    float* __restrict__ C, const float* __restrict__ bias,
    const float* __restrict__ Res, int resStride, int M, int N, int K)
{
  __shared__ __align__(16) unsigned short As[BM][LDSK];
  __shared__ __align__(16) unsigned short Bs[2][BN][LDSK];
  const int tid   = threadIdx.x;
  const int lane  = tid & 31;
  const int wave  = tid >> 5;
  const int waveM = wave >> 2;   // 0..1
  const int waveN = wave & 3;    // 0..3
  const int half  = lane >> 4;
  const int l15   = lane & 15;
  const int bm0   = blockIdx.y * BM;
  const int bn0   = blockIdx.x * BN;

  // --- per-thread staging addresses (fixed strides, branch-free) ---
  // A: row = tid>>2 (0..63), col group = (tid&3)*8  -> 8 f32 along K per thread
  const int arow = tid >> 2, acol = (tid & 3) << 3;
  const float* aptr = A + (size_t)(bm0 + arow) * K + acol;
  unsigned short* asd = &As[arow][acol];
  // B: n = tid>>1 (0..127) clamped to N-1 (OOB column only feeds unstored outputs),
  //    k half = (tid&1)*16 -> 16 bf16 (two b128 chunks) along K per thread
  const int brow = tid >> 1, bcol = (tid & 1) << 4;
  int gnrow = bn0 + brow; if (gnrow > N - 1) gnrow = N - 1;
  const unsigned short* bptr = Bw + (size_t)gnrow * K + bcol;
  const unsigned bofs0 = (unsigned)(size_t)&Bs[0][brow][bcol];
  const unsigned bofs1 = (unsigned)(size_t)&Bs[1][brow][bcol];

  v8f acc[2][2] = {};

  // --- prologue: async B slab0 -> Bs[0]; A slab0 -> regs ---
  async_copy_b128(bofs0,      bptr);
  async_copy_b128(bofs0 + 16, bptr + 8);
  float4 af0 = *(const float4*)(aptr);
  float4 af1 = *(const float4*)(aptr + 4);

  int buf = 0;
  for (int kb = 0; kb < K; kb += BK) {
    __syncthreads();                      // all waves done reading previous LDS slab
    // pack/store current A slab regs -> LDS
    uint4 apk;
    apk.x = f2bf_pk(af0.x, af0.y);
    apk.y = f2bf_pk(af0.z, af0.w);
    apk.z = f2bf_pk(af1.x, af1.y);
    apk.w = f2bf_pk(af1.z, af1.w);
    *(uint4*)asd = apk;
    wait_async0();                        // my async B copies for this slab landed
    __syncthreads();                      // publish As + Bs[buf]

    // prefetch next slab (clamped: last iter re-copies, discarded)
    const int kd = (kb + BK < K) ? BK : 0;
    af0 = *(const float4*)(aptr + kd);
    af1 = *(const float4*)(aptr + kd + 4);
    const unsigned bnext = buf ? bofs0 : bofs1;
    async_copy_b128(bnext,      bptr + kd);
    async_copy_b128(bnext + 16, bptr + kd + 8);

    // fragments per documented CDNA5 bf16 layouts; each = two ds_load_b128
    union FragU { v16bf v; uint4 q[2]; };
    FragU fa[2], fb[2];
    #pragma unroll
    for (int tm = 0; tm < 2; ++tm) {
      const unsigned short* ap = &As[waveM * 32 + tm * 16 + l15][8 * half];
      fa[tm].q[0] = *(const uint4*)(ap);        // K = 0..7   (+8 if half)
      fa[tm].q[1] = *(const uint4*)(ap + 16);   // K = 16..23 (+8 if half)
    }
    #pragma unroll
    for (int tn = 0; tn < 2; ++tn) {
      const unsigned short* bp = &Bs[buf][waveN * 32 + tn * 16 + l15][8 * half];
      fb[tn].q[0] = *(const uint4*)(bp);        // K = 0..7   (+8 if half)
      fb[tn].q[1] = *(const uint4*)(bp + 16);   // K = 16..23 (+8 if half)
    }
    #pragma unroll
    for (int tm = 0; tm < 2; ++tm)
      #pragma unroll
      for (int tn = 0; tn < 2; ++tn)
        acc[tm][tn] = __builtin_amdgcn_wmma_f32_16x16x32_bf16(
            false, fa[tm].v, false, fb[tn].v, (short)0, acc[tm][tn], false, false);

    aptr += BK;
    bptr += BK;
    buf ^= 1;
  }

  // Epilogue: C/D layout is M = r + 8*half, N = lane&15 per 16x16 tile
  #pragma unroll
  for (int tm = 0; tm < 2; ++tm) {
    const int gm = bm0 + waveM * 32 + tm * 16 + half * 8;
    #pragma unroll
    for (int tn = 0; tn < 2; ++tn) {
      const int gn = bn0 + waveN * 32 + tn * 16 + l15;
      if (gn < N) {
        const float bv = HAS_BIAS ? bias[gn] : 0.f;
        #pragma unroll
        for (int r = 0; r < 8; ++r) {
          float val = acc[tm][tn][r] + bv;
          if (GELU)    val = gelu_exact(val);
          if (HAS_RES) val += Res[(size_t)(gm + r) * resStride + gn];
          C[(size_t)(gm + r) * N + gn] = val;
        }
      }
    }
  }
}

// ---------------- orchestration ----------------

extern "C" void kernel_launch(void* const* d_in, const int* in_sizes, int n_in,
                              void* d_out, int out_size, void* d_ws, size_t ws_size,
                              hipStream_t stream) {
  (void)in_sizes; (void)n_in; (void)out_size; (void)ws_size;
  const int*   idx      = (const int*)  d_in[0];
  const int*   eidx     = (const int*)  d_in[1];
  const float* ew       = (const float*)d_in[2];
  const float* frozen   = (const float*)d_in[3];
  const float* W_root   = (const float*)d_in[4];
  const float* W_nbr    = (const float*)d_in[5];
  const float* b_mp     = (const float*)d_in[6];
  const float* W_post   = (const float*)d_in[7];
  const float* b_post   = (const float*)d_in[8];
  const float* fb_emb   = (const float*)d_in[9];
  const float* W_in     = (const float*)d_in[10];
  const float* b_in     = (const float*)d_in[11];
  const float* ln_g     = (const float*)d_in[12];
  const float* ln_b     = (const float*)d_in[13];
  const float* W1       = (const float*)d_in[14];
  const float* b1       = (const float*)d_in[15];
  const float* W2       = (const float*)d_in[16];
  const float* W_bil    = (const float*)d_in[17];
  const float* out_emb  = (const float*)d_in[18];
  float* out = (float*)d_out;

  char* ws = (char*)d_ws;
  size_t off = 0;
  auto alloc = [&](size_t bytes) -> void* {
    void* p = ws + off;
    off = (off + bytes + 255) & ~(size_t)255;
    return p;
  };
  int*   flags = (int*)  alloc((size_t)NNODES * 4);
  float* aggr  = (float*)alloc((size_t)NNODES * GDIM * 4);
  float* Acat  = (float*)alloc((size_t)BATCH * 2 * GDIM * 4);   // [Xg | Ag]
  float* T     = (float*)alloc((size_t)BATCH * GDIM * 4);
  float* P     = (float*)alloc((size_t)BATCH * GDIM * 4);
  float* H     = (float*)alloc((size_t)BATCH * HDIM * 4);
  float* XN    = (float*)alloc((size_t)BATCH * HDIM * 4);
  float* U     = (float*)alloc((size_t)BATCH * UDIM * 4);
  float* Proj  = (float*)alloc((size_t)BATCH * NCLS * HDIM * 4);
  // all weights stored bf16, N-major ([N,K])
  unsigned short* Wcat  = (unsigned short*)alloc((size_t)2 * GDIM * GDIM * 2);  // [256n, 512k]
  unsigned short* Wpost = (unsigned short*)alloc((size_t)GDIM * GDIM * 2);
  unsigned short* Win   = (unsigned short*)alloc((size_t)GDIM * HDIM * 2);
  unsigned short* W1b   = (unsigned short*)alloc((size_t)NRES * HDIM * UDIM * 2);
  unsigned short* W2b   = (unsigned short*)alloc((size_t)NRES * UDIM * HDIM * 2);
  unsigned short* Wbil  = (unsigned short*)alloc((size_t)HDIM * NCLS * HDIM * 2);
  unsigned short* OutT  = (unsigned short*)alloc((size_t)HDIM * NGENES * 2);    // out_emb is [N,K] already

  (void)hipMemsetAsync(flags, 0, (size_t)NNODES * 4, stream);
  (void)hipMemsetAsync(aggr,  0, (size_t)NNODES * GDIM * 4, stream);

  auto tr = [&](const float* s, unsigned short* d, int R, int C, int stride, int o) {
    cvt_tr_bf16<<<(R * C + 255) / 256, 256, 0, stream>>>(s, d, R, C, stride, o);
  };
  tr(W_root, Wcat, GDIM, GDIM, 2 * GDIM, 0);       // Wcat[n][0:256]   = W_root^T
  tr(W_nbr,  Wcat, GDIM, GDIM, 2 * GDIM, GDIM);    // Wcat[n][256:512] = W_nbr^T
  tr(W_post, Wpost, GDIM, GDIM, GDIM, 0);
  tr(W_in,   Win,   GDIM, HDIM, GDIM, 0);          // [512n, 256k]
  for (int i = 0; i < NRES; ++i) {
    tr(W1 + (size_t)i * HDIM * UDIM, W1b + (size_t)i * HDIM * UDIM, HDIM, UDIM, HDIM, 0);
    tr(W2 + (size_t)i * UDIM * HDIM, W2b + (size_t)i * UDIM * HDIM, UDIM, HDIM, UDIM, 0);
  }
  tr(W_bil, Wbil, HDIM, NCLS * HDIM, HDIM, 0);     // [1536n, 512k]
  cvt_f32_bf16<<<((HDIM * NGENES) + 255) / 256, 256, 0, stream>>>(out_emb, OutT,
                                                                  HDIM * NGENES);

  build_flags<<<(BATCH + 255) / 256, 256, 0, stream>>>(idx, flags);
  edge_scatter<<<(NEDGES + 7) / 8, 256, 0, stream>>>(eidx, ew, frozen, flags, aggr);
  gather_rows<<<(BATCH * 64) / 256, 256, 0, stream>>>(idx, frozen, aggr, Acat);

  // T = [Xg|Ag] @ [W_root;W_nbr] + b_mp + Xg
  {
    dim3 g(GDIM / BN, BATCH / BM);
    gemm_bf16_wmma<false, true, true><<<g, 256, 0, stream>>>(
        Acat, Wcat, T, b_mp, /*Res=*/Acat, /*resStride=*/2 * GDIM, BATCH, GDIM, 2 * GDIM);
  }
  // P = T @ W_post + b_post
  {
    dim3 g(GDIM / BN, BATCH / BM);
    gemm_bf16_wmma<false, true, false><<<g, 256, 0, stream>>>(
        T, Wpost, P, b_post, nullptr, 0, BATCH, GDIM, GDIM);
  }
  fallback_fixup<<<(BATCH * 64) / 256, 256, 0, stream>>>(idx, fb_emb, P);
  // H = gelu(P @ W_in + b_in)
  {
    dim3 g(HDIM / BN, BATCH / BM);
    gemm_bf16_wmma<true, true, false><<<g, 256, 0, stream>>>(
        P, Win, H, b_in, nullptr, 0, BATCH, HDIM, GDIM);
  }
  // residual MLP stack
  for (int i = 0; i < NRES; ++i) {
    layernorm512<<<BATCH / 8, 256, 0, stream>>>(H, XN, ln_g + i * HDIM, ln_b + i * HDIM);
    dim3 g1(UDIM / BN, BATCH / BM);
    gemm_bf16_wmma<true, true, false><<<g1, 256, 0, stream>>>(
        XN, W1b + (size_t)i * HDIM * UDIM, U, b1 + i * UDIM, nullptr, 0, BATCH, UDIM, HDIM);
    dim3 g2(HDIM / BN, BATCH / BM);
    gemm_bf16_wmma<false, false, true><<<g2, 256, 0, stream>>>(
        U, W2b + (size_t)i * UDIM * HDIM, H, nullptr, /*Res=*/H, /*resStride=*/HDIM,
        BATCH, HDIM, UDIM);
  }
  // Proj = H @ W_bil   [2048,1536]
  {
    dim3 g((NCLS * HDIM) / BN, BATCH / BM);
    gemm_bf16_wmma<false, false, false><<<g, 256, 0, stream>>>(
        H, Wbil, Proj, nullptr, nullptr, 0, BATCH, NCLS * HDIM, HDIM);
  }
  // logits = Proj(as [6144,512]) @ out_emb^T  -> [6144,6640] = d_out
  {
    dim3 g((NGENES + BN - 1) / BN, (BATCH * NCLS) / BM);
    gemm_bf16_wmma<false, false, false><<<g, 256, 0, stream>>>(
        Proj, OutT, out, nullptr, nullptr, 0, BATCH * NCLS, NGENES, HDIM);
  }
}
